// MoE_1331439862381
// MI455X (gfx1250) — matrix operand order repaired and checked by reading, not compiled, not run on hardware
//
#include <hip/hip_runtime.h>
#include <math.h>

// ---------------- problem constants ----------------
#define Hdim 1024
#define Fdim 4096
#define NTOK 8192          // 4 * 2048 tokens
#define EROUTED 8

#define TILE 128
#define KSTEP 32
#define APITCH 40          // bf16 elems per LDS row: 80B stride = 20 banks, 16B aligned
#define ABUF (TILE * APITCH)   // elements per LDS buffer (10240 B)

typedef __attribute__((ext_vector_type(16))) __bf16 v16bf;
typedef __attribute__((ext_vector_type(8)))  __bf16 v8bf;
typedef __attribute__((ext_vector_type(4)))  __bf16 v4bf;
typedef __attribute__((ext_vector_type(8)))  float  v8f;
typedef __attribute__((ext_vector_type(4)))  int    v4i;

// --------- CDNA5 async global->LDS path (guarded; reg-staged fallback) ---------
#if defined(__has_builtin)
#if __has_builtin(__builtin_amdgcn_global_load_async_to_lds_b128)
#define HAVE_ASYNC_LDS 1
#endif
#endif
#ifndef HAVE_ASYNC_LDS
#define HAVE_ASYNC_LDS 0
#endif

__device__ __forceinline__ void wait_async0() {
#if defined(__has_builtin) && __has_builtin(__builtin_amdgcn_s_wait_asynccnt)
  __builtin_amdgcn_s_wait_asynccnt(0);
#else
  asm volatile("s_wait_asynccnt 0" ::: "memory");
#endif
}

__device__ __forceinline__ v16bf frag_cat(v8bf lo, v8bf hi) {
  return __builtin_shufflevector(lo, hi, 0,1,2,3,4,5,6,7,8,9,10,11,12,13,14,15);
}

// ---------------- router: softmax + top-2 gates ----------------
__global__ __launch_bounds__(256)
void router_kernel(const float* __restrict__ x, const float* __restrict__ rw,
                   float* __restrict__ gates) {
  const int lane = threadIdx.x & 31;
  const int wid  = threadIdx.x >> 5;
  const int token = blockIdx.x * 8 + wid;
  const float* xr = x + (size_t)token * Hdim;

  float acc[EROUTED];
#pragma unroll
  for (int e = 0; e < EROUTED; e++) acc[e] = 0.f;

  for (int k = lane; k < Hdim; k += 32) {
    float xv = xr[k];
    const float* w = rw + (size_t)k * EROUTED;
#pragma unroll
    for (int e = 0; e < EROUTED; e++) acc[e] = fmaf(xv, w[e], acc[e]);
  }
#pragma unroll
  for (int off = 16; off > 0; off >>= 1) {
#pragma unroll
    for (int e = 0; e < EROUTED; e++) acc[e] += __shfl_xor(acc[e], off, 32);
  }
  float mx = acc[0];
#pragma unroll
  for (int e = 1; e < EROUTED; e++) mx = fmaxf(mx, acc[e]);
  float p[EROUTED]; float s = 0.f;
#pragma unroll
  for (int e = 0; e < EROUTED; e++) { p[e] = expf(acc[e] - mx); s += p[e]; }
  float inv = 1.0f / s;
#pragma unroll
  for (int e = 0; e < EROUTED; e++) p[e] *= inv;
  int i1 = 0;
#pragma unroll
  for (int e = 1; e < EROUTED; e++) if (p[e] > p[i1]) i1 = e;
  int i2 = (i1 == 0) ? 1 : 0;
#pragma unroll
  for (int e = 0; e < EROUTED; e++) if (e != i1 && p[e] > p[i2]) i2 = e;

  if (lane < EROUTED)
    gates[(size_t)token * EROUTED + lane] = (lane == i1 || lane == i2) ? p[lane] : 0.f;
}

// ---------------- fp32 -> bf16 cast (x) ----------------
__global__ __launch_bounds__(256)
void cast_bf16_kernel(const float* __restrict__ in, __bf16* __restrict__ out, int n4) {
  int i = blockIdx.x * blockDim.x + threadIdx.x;
  if (i >= n4) return;
  float4 v = ((const float4*)in)[i];
  v4bf o;
  o[0] = (__bf16)v.x; o[1] = (__bf16)v.y; o[2] = (__bf16)v.z; o[3] = (__bf16)v.w;
  ((v4bf*)out)[i] = o;
}

// ---------------- fused WMMA GEMM (double-buffered LDS) ----------------
// C[M][N] = A[M][K](bf16) @ B[K][N](fp32 -> bf16 in LDS)
// IS_UP: epilogue = exact gelu, bf16 h staged through LDS for coalesced stores
// !IS_UP: epilogue = out = x + acc (eidx<0) or atomicAdd(out, gate*acc)
template <bool IS_UP>
__global__ __launch_bounds__(256)
void moe_gemm_kernel(const __bf16* __restrict__ A, const float* __restrict__ B,
                     int K, int N,
                     __bf16* __restrict__ Hout,
                     const float* __restrict__ xres,
                     const float* __restrict__ gates, int eidx,
                     float* __restrict__ out) {
  __shared__ __bf16 As[2 * ABUF];
  __shared__ __bf16 Bs[2 * ABUF];
  __shared__ __bf16 Cs[IS_UP ? (TILE * TILE) : 1];

  const int tid  = threadIdx.x;
  const int lane = tid & 31;
  const int wid  = tid >> 5;
  const int half = lane >> 4;
  const int lm   = lane & 15;
  const int wm   = wid & 1;            // 2 waves along M (64 rows)
  const int wn   = wid >> 1;           // 4 waves along N (32 cols)
  const int m0   = blockIdx.y * TILE;
  const int n0   = blockIdx.x * TILE;

  // ---- hoisted staging maps ----
  // A: 2 x b128 chunks per thread (128 rows x 32 bf16)
  int aLds[2]; const __bf16* aPtr[2];
#pragma unroll
  for (int i = 0; i < 2; i++) {
    int idx = i * 256 + tid;
    int r = idx >> 2, c = (idx & 3) * 8;
    aLds[i] = r * APITCH + c;
    aPtr[i] = A + (size_t)(m0 + r) * K + c;
  }
  // B: 2 chunks per thread; each = 2 adjacent K rows x 4 floats (packs bf16 pairs)
  int br2[2], bc4[2]; const float* bPtr[2];
#pragma unroll
  for (int i = 0; i < 2; i++) {
    int idx = i * 256 + tid;            // 512 chunks: 16 k-pairs x 32 col-groups
    br2[i] = idx >> 5; bc4[i] = idx & 31;
    bPtr[i] = B + (size_t)(2 * br2[i]) * N + n0 + bc4[i] * 4;
  }
  // compute-side fragment offsets
  int aOff[4], bOff[2];
#pragma unroll
  for (int mt = 0; mt < 4; mt++) aOff[mt] = (wm * 64 + mt * 16 + lm) * APITCH + half * 8;
#pragma unroll
  for (int nt = 0; nt < 2; nt++) bOff[nt] = (wn * 32 + nt * 16 + lm) * APITCH + half * 8;

#if !HAVE_ASYNC_LDS
  uint4 aReg[2];
#endif
  float4 bLo[2], bHi[2];

  auto issueA = [&](int buf) {           // start A(k) -> As[buf]
#pragma unroll
    for (int i = 0; i < 2; i++) {
#if HAVE_ASYNC_LDS
      __builtin_amdgcn_global_load_async_to_lds_b128(
          (v4i*)aPtr[i],
          (v4i*)&As[buf * ABUF + aLds[i]],
          0, 0);
#else
      aReg[i] = *(const uint4*)aPtr[i];
#endif
      aPtr[i] += KSTEP;
    }
  };
  auto loadB = [&]() {
#pragma unroll
    for (int i = 0; i < 2; i++) {
      bLo[i] = *(const float4*)bPtr[i];
      bHi[i] = *(const float4*)(bPtr[i] + N);
      bPtr[i] += (size_t)KSTEP * N;
    }
  };
  auto landA = [&](int buf) {            // make A(k) resident in As[buf]
#if HAVE_ASYNC_LDS
    (void)buf;
    wait_async0();
#else
#pragma unroll
    for (int i = 0; i < 2; i++)
      *(uint4*)(&As[buf * ABUF + aLds[i]]) = aReg[i];
#endif
  };
  auto landB = [&](int buf) {            // fp32->bf16, transpose, packed 4B stores
#pragma unroll
    for (int i = 0; i < 2; i++) {
      const float* lo = (const float*)&bLo[i];
      const float* hi = (const float*)&bHi[i];
#pragma unroll
      for (int j = 0; j < 4; j++) {
        int f = bc4[i] * 4 + j;
        union { __bf16 h[2]; unsigned u; } pk;
        pk.h[0] = (__bf16)lo[j];         // k = 2*br2
        pk.h[1] = (__bf16)hi[j];         // k = 2*br2 + 1
        *(unsigned*)&Bs[buf * ABUF + f * APITCH + 2 * br2[i]] = pk.u;
      }
    }
  };

  v8f acc[4][2];
#pragma unroll
  for (int mt = 0; mt < 4; mt++)
#pragma unroll
    for (int nt = 0; nt < 2; nt++)
#pragma unroll
      for (int r = 0; r < 8; r++) acc[mt][nt][r] = 0.f;

  issueA(0);
  loadB();

  int cur = 0;
  for (int k0 = 0; k0 < K; k0 += KSTEP) {
    landA(cur);
    landB(cur);
    __syncthreads();                     // tile(cur) visible to all waves
    if (k0 + KSTEP < K) { issueA(cur ^ 1); loadB(); }
    if (k0 + 2 * KSTEP < K) {            // prefetch weight rows two tiles ahead
      __builtin_prefetch(bPtr[0], 0, 1);
      __builtin_prefetch(bPtr[1], 0, 1);
    }

    v16bf af[4], bfr[2];
#pragma unroll
    for (int mt = 0; mt < 4; mt++) {
      int base = cur * ABUF + aOff[mt];
      v8bf lo = *(const v8bf*)(&As[base]);
      v8bf hi = *(const v8bf*)(&As[base + 16]);
      af[mt] = frag_cat(lo, hi);
    }
#pragma unroll
    for (int nt = 0; nt < 2; nt++) {
      int base = cur * ABUF + bOff[nt];
      v8bf lo = *(const v8bf*)(&Bs[base]);
      v8bf hi = *(const v8bf*)(&Bs[base + 16]);
      bfr[nt] = frag_cat(lo, hi);
    }
#pragma unroll
    for (int mt = 0; mt < 4; mt++)
#pragma unroll
      for (int nt = 0; nt < 2; nt++)
        acc[mt][nt] = __builtin_amdgcn_wmma_f32_16x16x32_bf16(
            false, af[mt], false, bfr[nt], (short)0, acc[mt][nt], false, false);

    cur ^= 1;
  }

  if (IS_UP) {
    __syncthreads();
#pragma unroll
    for (int mt = 0; mt < 4; mt++)
#pragma unroll
      for (int nt = 0; nt < 2; nt++)
#pragma unroll
        for (int r = 0; r < 8; r++) {
          float v = acc[mt][nt][r];
          float g = 0.5f * v * (1.0f + erff(v * 0.70710678118654752f));
          int row = wm * 64 + mt * 16 + half * 8 + r;  // C/D: lanes>=16 -> M+8
          int col = wn * 32 + nt * 16 + lm;
          Cs[row * TILE + col] = (__bf16)g;
        }
    __syncthreads();
#pragma unroll
    for (int i = 0; i < 8; i++) {
      int idx = i * 256 + tid;                         // 2048 x uint4
      int r = idx >> 4, c = idx & 15;
      *(uint4*)(Hout + (size_t)(m0 + r) * N + n0 + c * 8) =
          *(const uint4*)(&Cs[r * TILE + c * 8]);
    }
  } else {
#pragma unroll
    for (int mt = 0; mt < 4; mt++)
#pragma unroll
      for (int r = 0; r < 8; r++) {
        int row = wm * 64 + mt * 16 + half * 8 + r;
        int token = m0 + row;
        float gv = (eidx < 0) ? 1.0f : gates[(size_t)token * EROUTED + eidx];
#pragma unroll
        for (int nt = 0; nt < 2; nt++) {
          int col = n0 + wn * 32 + nt * 16 + lm;
          size_t o = (size_t)token * N + col;
          float v = acc[mt][nt][r] * gv;
          if (eidx < 0) out[o] = xres[o] + v;          // first pass: out = x + shared
          else          atomicAdd(&out[o], v);         // no-return f32 atomic at L2
        }
      }
  }
}

// ---------------- host launcher ----------------
extern "C" void kernel_launch(void* const* d_in, const int* in_sizes, int n_in,
                              void* d_out, int out_size, void* d_ws, size_t ws_size,
                              hipStream_t stream) {
  const float* x           = (const float*)d_in[0];
  const float* shared_up   = (const float*)d_in[1];
  const float* shared_down = (const float*)d_in[2];
  const float* routed_up   = (const float*)d_in[3];
  const float* routed_down = (const float*)d_in[4];
  const float* router_w    = (const float*)d_in[5];
  float* out = (float*)d_out;

  char* ws = (char*)d_ws;
  const size_t GATES_BYTES = (size_t)NTOK * EROUTED * sizeof(float);
  const size_t XB_BYTES    = (size_t)NTOK * Hdim * sizeof(__bf16);
  float*  gates = (float*)ws;
  __bf16* xb    = (__bf16*)(ws + GATES_BYTES);
  __bf16* hbuf  = (__bf16*)(ws + GATES_BYTES + XB_BYTES);

  router_kernel<<<NTOK / 8, 256, 0, stream>>>(x, router_w, gates);
  cast_bf16_kernel<<<(NTOK * Hdim / 4 + 255) / 256, 256, 0, stream>>>(x, xb, NTOK * Hdim / 4);

  dim3 upGrid(Fdim / TILE, NTOK / TILE);   // 32 x 64
  dim3 dnGrid(Hdim / TILE, NTOK / TILE);   //  8 x 64

  moe_gemm_kernel<true><<<upGrid, 256, 0, stream>>>(
      xb, shared_up, Hdim, Fdim, hbuf, nullptr, nullptr, -1, nullptr);
  moe_gemm_kernel<false><<<dnGrid, 256, 0, stream>>>(
      hbuf, shared_down, Fdim, Hdim, nullptr, x, gates, -1, out);

  for (int e = 0; e < EROUTED; e++) {
    moe_gemm_kernel<true><<<upGrid, 256, 0, stream>>>(
        xb, routed_up + (size_t)e * Hdim * Fdim, Hdim, Fdim, hbuf,
        nullptr, nullptr, -1, nullptr);
    moe_gemm_kernel<false><<<dnGrid, 256, 0, stream>>>(
        hbuf, routed_down + (size_t)e * Fdim * Hdim, Fdim, Hdim, nullptr,
        x, gates, e, out);
  }
}
